// PointTransformerBlock_40596030881802
// MI455X (gfx1250) — matrix operand order repaired and checked by reading, not compiled, not run on hardware
//
#include <hip/hip_runtime.h>
#include <math.h>

// ---------------- problem constants ----------------
#define B_    4
#define N_    4096          // 2^12
#define D_    256
#define H_    4
#define KNN_  16
#define HID_  64
#define FF_   1024
#define BN_   (B_ * N_)     // 16384
#define ROWS_ (BN_ * KNN_)  // 262144

// ---------------- bf16 helpers ----------------
typedef __attribute__((ext_vector_type(16))) __bf16 bf16x16;
typedef __attribute__((ext_vector_type(8)))  float  f32x8;

union Frag {
  bf16x16      v;
  unsigned int u[8];
  uint4        q[2];
};

__device__ __forceinline__ unsigned short f2bf(float f) {
  unsigned int u = __float_as_uint(f);
  u = u + 0x7FFFu + ((u >> 16) & 1u);           // round-to-nearest-even
  return (unsigned short)(u >> 16);
}
__device__ __forceinline__ float bf2f(unsigned short s) {
  return __uint_as_float(((unsigned int)s) << 16);
}
__device__ __forceinline__ float gelu_exact(float x) {
  return 0.5f * x * (1.0f + erff(x * 0.70710678118654752f));
}
__device__ __forceinline__ float wave_sum(float v) {
  #pragma unroll
  for (int m = 16; m >= 1; m >>= 1) v += __shfl_xor(v, m, 32);
  return v;
}

// ---------------- 1) KNN (top-16 nearest, self excluded) ----------------
__global__ __launch_bounds__(256) void knn_kernel(const float* __restrict__ xyz,
                                                  int* __restrict__ idx) {
  int t = blockIdx.x * blockDim.x + threadIdx.x;       // over B*N
  if (t >= BN_) return;
  int b = t >> 12, n = t & (N_ - 1);
  const float* base = xyz + (size_t)b * N_ * 3;
  float qx = base[n * 3 + 0], qy = base[n * 3 + 1], qz = base[n * 3 + 2];

  float bd[KNN_]; int bi[KNN_];
  #pragma unroll
  for (int i = 0; i < KNN_; i++) { bd[i] = 1e30f; bi[i] = 0; }
  float worst = 1e30f; int wpos = 0;

  for (int m = 0; m < N_; m++) {
    if (m == n) continue;
    float dx = base[m * 3 + 0] - qx;
    float dy = base[m * 3 + 1] - qy;
    float dz = base[m * 3 + 2] - qz;
    float d2 = dx * dx + dy * dy + dz * dz;
    if (d2 < worst) {
      #pragma unroll
      for (int i = 0; i < KNN_; i++) if (i == wpos) { bd[i] = d2; bi[i] = m; }
      float mx = -1.0f; int mp = 0;
      #pragma unroll
      for (int i = 0; i < KNN_; i++) if (bd[i] > mx) { mx = bd[i]; mp = i; }
      worst = mx; wpos = mp;
    }
  }
  #pragma unroll
  for (int i = 0; i < KNN_; i++) idx[(size_t)t * KNN_ + i] = bi[i];
}

// ---------------- 2) LayerNorm -> bf16 (wave per 256-elem row) ----------------
__global__ __launch_bounds__(256) void ln_kernel(const float* __restrict__ x,
                                                 const float* __restrict__ g,
                                                 const float* __restrict__ bta,
                                                 unsigned short* __restrict__ out,
                                                 int rows) {
  int w = blockIdx.x * 8 + (threadIdx.x >> 5);
  int lane = threadIdx.x & 31;
  if (w >= rows) return;
  const float* row = x + (size_t)w * D_;
  float vals[8], s = 0.f;
  #pragma unroll
  for (int i = 0; i < 8; i++) { vals[i] = row[lane + 32 * i]; s += vals[i]; }
  s = wave_sum(s);
  float mu = s * (1.0f / D_);
  float s2 = 0.f;
  #pragma unroll
  for (int i = 0; i < 8; i++) { float d = vals[i] - mu; s2 += d * d; }
  s2 = wave_sum(s2);
  float rinv = rsqrtf(s2 * (1.0f / D_) + 1e-5f);
  #pragma unroll
  for (int i = 0; i < 8; i++) {
    int c = lane + 32 * i;
    out[(size_t)w * D_ + c] = f2bf((vals[i] - mu) * rinv * g[c] + bta[c]);
  }
}

// ---------------- 3) weight convert + transpose: Wt[n*K+k] = bf16(W[k*N+n]) ----------------
__global__ __launch_bounds__(256) void conv_t_kernel(const float* __restrict__ W,
                                                     unsigned short* __restrict__ Wt,
                                                     int Kd, int Nd) {
  int i = blockIdx.x * blockDim.x + threadIdx.x;
  if (i >= Kd * Nd) return;
  int nn = i % Nd, kk = i / Nd;
  Wt[(size_t)nn * Kd + kk] = f2bf(W[i]);
}

// ---------------- 4) WMMA GEMM ----------------
// C[M,N] = A[M,K](bf16) * Bt[N,K](bf16)^T + bias, wave computes a 32x64 tile:
// 2 A-frags x 4 B-frags = 8 v_wmma_f32_16x16x32_bf16 per 32-wide K step, with
// explicit K double-buffering so next-step b128 loads are in flight during WMMAs.
// Requires: M % 32 == 0, N % 64 == 0, K % 64 == 0 (true for all call sites).

__device__ __forceinline__ void load_frag_a(Frag& f, const unsigned short* row,
                                            int k0, int half) {
  // 16-bit A 16x32 layout: V0..3 <- K = 8*half + [0,8); V4..7 <- K = 16 + 8*half + [0,8)
  f.q[0] = *(const uint4*)(row + k0 + half * 8);
  f.q[1] = *(const uint4*)(row + k0 + 16 + half * 8);
}
__device__ __forceinline__ void load_frag_b(Frag& f, const unsigned short* row,
                                            int k0, int half) {
  // 16-bit B 32x16 layout (lane = column): V0..3 <- K = 8*half + [0,8); V4..7 <- K = 16 + 8*half + [0,8)
  f.q[0] = *(const uint4*)(row + k0 + half * 8);
  f.q[1] = *(const uint4*)(row + k0 + 16 + half * 8);
}

struct KTile { Frag a0, a1, b0, b1, b2, b3; };

__device__ __forceinline__ void load_ktile(KTile& T,
    const unsigned short* ar0, const unsigned short* ar1,
    const unsigned short* br0, const unsigned short* br1,
    const unsigned short* br2, const unsigned short* br3,
    int k0, int half) {
  load_frag_a(T.a0, ar0, k0, half);
  load_frag_a(T.a1, ar1, k0, half);
  load_frag_b(T.b0, br0, k0, half);
  load_frag_b(T.b1, br1, k0, half);
  load_frag_b(T.b2, br2, k0, half);
  load_frag_b(T.b3, br3, k0, half);
}

__device__ __forceinline__ void mma_ktile(const KTile& T, f32x8 (&acc)[2][4]) {
  acc[0][0] = __builtin_amdgcn_wmma_f32_16x16x32_bf16(false, T.a0.v, false, T.b0.v, (short)0, acc[0][0], false, false);
  acc[0][1] = __builtin_amdgcn_wmma_f32_16x16x32_bf16(false, T.a0.v, false, T.b1.v, (short)0, acc[0][1], false, false);
  acc[0][2] = __builtin_amdgcn_wmma_f32_16x16x32_bf16(false, T.a0.v, false, T.b2.v, (short)0, acc[0][2], false, false);
  acc[0][3] = __builtin_amdgcn_wmma_f32_16x16x32_bf16(false, T.a0.v, false, T.b3.v, (short)0, acc[0][3], false, false);
  acc[1][0] = __builtin_amdgcn_wmma_f32_16x16x32_bf16(false, T.a1.v, false, T.b0.v, (short)0, acc[1][0], false, false);
  acc[1][1] = __builtin_amdgcn_wmma_f32_16x16x32_bf16(false, T.a1.v, false, T.b1.v, (short)0, acc[1][1], false, false);
  acc[1][2] = __builtin_amdgcn_wmma_f32_16x16x32_bf16(false, T.a1.v, false, T.b2.v, (short)0, acc[1][2], false, false);
  acc[1][3] = __builtin_amdgcn_wmma_f32_16x16x32_bf16(false, T.a1.v, false, T.b3.v, (short)0, acc[1][3], false, false);
}

template <bool GELU, bool RES>
__global__ __launch_bounds__(256) void wmma_gemm_kernel(
    const unsigned short* __restrict__ A, const unsigned short* __restrict__ Bt,
    const float* __restrict__ bias, const float* __restrict__ res,
    float* __restrict__ outF, unsigned short* __restrict__ outH,
    int M, int Nd, int Kd) {
  const int w    = blockIdx.x * 8 + (threadIdx.x >> 5);
  const int lane = threadIdx.x & 31;
  const int mtiles = M >> 5;          // 32-row tiles
  const int n64    = Nd >> 6;
  if (w >= mtiles * n64) return;
  const int mt   = w % mtiles;
  const int nt   = w / mtiles;
  const int half = lane >> 4;
  const int l16  = lane & 15;

  const unsigned short* ar0 = A + (size_t)((mt << 5) + l16) * Kd;
  const unsigned short* ar1 = A + (size_t)((mt << 5) + 16 + l16) * Kd;
  const unsigned short* br0 = Bt + (size_t)((nt << 6) + 0 * 16 + l16) * Kd;
  const unsigned short* br1 = Bt + (size_t)((nt << 6) + 1 * 16 + l16) * Kd;
  const unsigned short* br2 = Bt + (size_t)((nt << 6) + 2 * 16 + l16) * Kd;
  const unsigned short* br3 = Bt + (size_t)((nt << 6) + 3 * 16 + l16) * Kd;

  f32x8 acc[2][4] = {};
  KTile T0, T1;
  load_ktile(T0, ar0, ar1, br0, br1, br2, br3, 0, half);
  for (int k0 = 0; k0 < Kd; k0 += 64) {
    // prefetch second half of this 64-wide super-step while T0 is consumed
    load_ktile(T1, ar0, ar1, br0, br1, br2, br3, k0 + 32, half);
    mma_ktile(T0, acc);
    if (k0 + 64 < Kd)
      load_ktile(T0, ar0, ar1, br0, br1, br2, br3, k0 + 64, half);
    mma_ktile(T1, acc);
  }

  // Epilogue: C/D layout — lane l16 = column, VGPR i = row i (lanes 0-15) / 8+i (lanes 16-31)
  #pragma unroll
  for (int mi = 0; mi < 2; mi++) {
    #pragma unroll
    for (int t = 0; t < 4; t++) {
      const int ncol = (nt << 6) + t * 16 + l16;
      const float bia = bias[ncol];
      #pragma unroll
      for (int i = 0; i < 8; i++) {
        const int row = (mt << 5) + mi * 16 + half * 8 + i;
        float val = acc[mi][t][i] + bia;
        if (GELU) val = gelu_exact(val);
        if (RES)  val += res[(size_t)row * Nd + ncol];
        if (outF) outF[(size_t)row * Nd + ncol] = val;
        if (outH) outH[(size_t)row * Nd + ncol] = f2bf(val);
      }
    }
  }
}

// ---------------- 5) rel-pos MLP hidden: gelu(rel(4) @ W_rel1 + b_rel1) -> bf16 ----------------
__global__ __launch_bounds__(256) void rel_hidden_kernel(const float* __restrict__ xyz,
                                                         const int* __restrict__ idx,
                                                         const float* __restrict__ W1,
                                                         const float* __restrict__ b1,
                                                         unsigned short* __restrict__ hid) {
  int w = blockIdx.x * 8 + (threadIdx.x >> 5);   // over B*N*KNN rows
  int lane = threadIdx.x & 31;
  if (w >= ROWS_) return;
  int j = w & (KNN_ - 1), bn = w >> 4;
  int b = bn >> 12, n = bn & (N_ - 1);
  int nb = idx[(size_t)bn * KNN_ + j];
  const float* p0 = xyz + ((size_t)b * N_ + n) * 3;
  const float* p1 = xyz + ((size_t)b * N_ + nb) * 3;
  float dx = p1[0] - p0[0], dy = p1[1] - p0[1], dz = p1[2] - p0[2];
  float dn = sqrtf(dx * dx + dy * dy + dz * dz);
  #pragma unroll
  for (int t = 0; t < 2; t++) {
    int o = 2 * lane + t;
    float hv = b1[o] + dx * W1[0 * HID_ + o] + dy * W1[1 * HID_ + o] +
               dz * W1[2 * HID_ + o] + dn * W1[3 * HID_ + o];
    hid[(size_t)w * HID_ + o] = f2bf(gelu_exact(hv));
  }
}

// ---------------- 6) attention: wave per (b,n,h); k=16 neighbors, dh=64 ----------------
__global__ __launch_bounds__(256) void attn_kernel(const unsigned short* __restrict__ qb,
                                                   const unsigned short* __restrict__ kb,
                                                   const unsigned short* __restrict__ vb,
                                                   const unsigned short* __restrict__ rel,
                                                   const int* __restrict__ idx,
                                                   unsigned short* __restrict__ out) {
  int w = blockIdx.x * 8 + (threadIdx.x >> 5);   // = (b*N+n)*H + h
  int lane = threadIdx.x & 31;
  if (w >= BN_ * H_) return;
  int h = w & (H_ - 1);
  int bn = w >> 2;
  int b = bn >> 12;
  int c0 = 2 * lane, c1 = 2 * lane + 1;
  size_t qoff = (size_t)bn * D_ + h * 64;
  float q0 = bf2f(qb[qoff + c0]), q1 = bf2f(qb[qoff + c1]);

  float sc[KNN_], r0[KNN_], r1[KNN_];
  int nb[KNN_];
  #pragma unroll
  for (int j = 0; j < KNN_; j++) {
    nb[j] = idx[(size_t)bn * KNN_ + j];
    size_t krow = ((size_t)b * N_ + nb[j]) * D_ + h * 64;
    size_t rrow = ((size_t)bn * KNN_ + j) * D_ + h * 64;
    r0[j] = bf2f(rel[rrow + c0]);
    r1[j] = bf2f(rel[rrow + c1]);
    float k0v = bf2f(kb[krow + c0]) + r0[j];
    float k1v = bf2f(kb[krow + c1]) + r1[j];
    sc[j] = wave_sum(q0 * k0v + q1 * k1v) * 0.125f;   // 1/sqrt(64)
  }
  float mx = sc[0];
  #pragma unroll
  for (int j = 1; j < KNN_; j++) mx = fmaxf(mx, sc[j]);
  float sum = 0.f;
  #pragma unroll
  for (int j = 0; j < KNN_; j++) { sc[j] = expf(sc[j] - mx); sum += sc[j]; }
  float inv = 1.0f / sum;
  float a0 = 0.f, a1 = 0.f;
  #pragma unroll
  for (int j = 0; j < KNN_; j++) {
    size_t vrow = ((size_t)b * N_ + nb[j]) * D_ + h * 64;
    a0 += sc[j] * (bf2f(vb[vrow + c0]) + r0[j]);
    a1 += sc[j] * (bf2f(vb[vrow + c1]) + r1[j]);
  }
  out[qoff + c0] = f2bf(a0 * inv);
  out[qoff + c1] = f2bf(a1 * inv);
}

// ---------------- host orchestration ----------------
extern "C" void kernel_launch(void* const* d_in, const int* in_sizes, int n_in,
                              void* d_out, int out_size, void* d_ws, size_t ws_size,
                              hipStream_t stream) {
  const float* xyz    = (const float*)d_in[0];
  const float* feats  = (const float*)d_in[1];
  const float* ln_q_g = (const float*)d_in[2];
  const float* ln_q_b = (const float*)d_in[3];
  const float* Wq = (const float*)d_in[4];  const float* bq = (const float*)d_in[5];
  const float* Wk = (const float*)d_in[6];  const float* bk = (const float*)d_in[7];
  const float* Wv = (const float*)d_in[8];  const float* bv = (const float*)d_in[9];
  const float* W_rel1 = (const float*)d_in[10]; const float* b_rel1 = (const float*)d_in[11];
  const float* W_rel2 = (const float*)d_in[12]; const float* b_rel2 = (const float*)d_in[13];
  const float* Wo = (const float*)d_in[14]; const float* bo = (const float*)d_in[15];
  const float* ln_f_g = (const float*)d_in[16]; const float* ln_f_b = (const float*)d_in[17];
  const float* W_ff1 = (const float*)d_in[18]; const float* b_ff1 = (const float*)d_in[19];
  const float* W_ff2 = (const float*)d_in[20]; const float* b_ff2 = (const float*)d_in[21];
  float* outp = (float*)d_out;

  char* ws = (char*)d_ws;
  size_t off = 0;
  auto alloc = [&](size_t bytes) { size_t o = off; off = (off + bytes + 255) & ~(size_t)255; return o; };

  int*            idxk   = (int*)           (ws + alloc(sizeof(int)   * (size_t)BN_ * KNN_));
  unsigned short* x_ln   = (unsigned short*)(ws + alloc(sizeof(short) * (size_t)BN_ * D_));
  unsigned short* qb     = (unsigned short*)(ws + alloc(sizeof(short) * (size_t)BN_ * D_));
  unsigned short* kbuf   = (unsigned short*)(ws + alloc(sizeof(short) * (size_t)BN_ * D_));
  unsigned short* vbuf   = (unsigned short*)(ws + alloc(sizeof(short) * (size_t)BN_ * D_));
  unsigned short* hid    = (unsigned short*)(ws + alloc(sizeof(short) * (size_t)ROWS_ * HID_));
  unsigned short* rel_e  = (unsigned short*)(ws + alloc(sizeof(short) * (size_t)ROWS_ * D_));
  unsigned short* attnb  = (unsigned short*)(ws + alloc(sizeof(short) * (size_t)BN_ * D_));
  float*          feats2 = (float*)         (ws + alloc(sizeof(float) * (size_t)BN_ * D_));
  unsigned short* h_ln   = (unsigned short*)(ws + alloc(sizeof(short) * (size_t)BN_ * D_));
  unsigned short* ff1    = (unsigned short*)(ws + alloc(sizeof(short) * (size_t)BN_ * FF_));
  unsigned short* Wq_t   = (unsigned short*)(ws + alloc(sizeof(short) * (size_t)D_ * D_));
  unsigned short* Wk_t   = (unsigned short*)(ws + alloc(sizeof(short) * (size_t)D_ * D_));
  unsigned short* Wv_t   = (unsigned short*)(ws + alloc(sizeof(short) * (size_t)D_ * D_));
  unsigned short* Wo_t   = (unsigned short*)(ws + alloc(sizeof(short) * (size_t)D_ * D_));
  unsigned short* Wr2_t  = (unsigned short*)(ws + alloc(sizeof(short) * (size_t)HID_ * D_));
  unsigned short* Wf1_t  = (unsigned short*)(ws + alloc(sizeof(short) * (size_t)D_ * FF_));
  unsigned short* Wf2_t  = (unsigned short*)(ws + alloc(sizeof(short) * (size_t)FF_ * D_));

  // 1) KNN
  knn_kernel<<<BN_ / 256, 256, 0, stream>>>(xyz, idxk);

  // 2) LN(feats) -> x_ln
  ln_kernel<<<BN_ / 8, 256, 0, stream>>>(feats, ln_q_g, ln_q_b, x_ln, BN_);

  // 3) weight conversions (bf16, transposed to N x K)
  auto conv = [&](const float* W, unsigned short* Wt, int Kd, int Nd) {
    int e = Kd * Nd;
    conv_t_kernel<<<(e + 255) / 256, 256, 0, stream>>>(W, Wt, Kd, Nd);
  };
  conv(Wq, Wq_t, D_, D_);
  conv(Wk, Wk_t, D_, D_);
  conv(Wv, Wv_t, D_, D_);
  conv(Wo, Wo_t, D_, D_);
  conv(W_rel2, Wr2_t, HID_, D_);
  conv(W_ff1, Wf1_t, D_, FF_);
  conv(W_ff2, Wf2_t, FF_, D_);

  auto gemm = [&](auto kern, const unsigned short* A, const unsigned short* Bt,
                  const float* bias, const float* res, float* oF, unsigned short* oH,
                  int M, int Nd, int Kd) {
    int waves = (M >> 5) * (Nd >> 6);
    kern<<<(waves + 7) / 8, 256, 0, stream>>>(A, Bt, bias, res, oF, oH, M, Nd, Kd);
  };

  // 4) Q/K/V projections
  gemm(wmma_gemm_kernel<false, false>, x_ln, Wq_t, bq, nullptr, nullptr, qb,   BN_, D_, D_);
  gemm(wmma_gemm_kernel<false, false>, x_ln, Wk_t, bk, nullptr, nullptr, kbuf, BN_, D_, D_);
  gemm(wmma_gemm_kernel<false, false>, x_ln, Wv_t, bv, nullptr, nullptr, vbuf, BN_, D_, D_);

  // 5) relative-position MLP
  rel_hidden_kernel<<<ROWS_ / 8, 256, 0, stream>>>(xyz, idxk, W_rel1, b_rel1, hid);
  gemm(wmma_gemm_kernel<false, false>, hid, Wr2_t, b_rel2, nullptr, nullptr, rel_e, ROWS_, D_, HID_);

  // 6) attention
  attn_kernel<<<(BN_ * H_) / 8, 256, 0, stream>>>(qb, kbuf, vbuf, rel_e, idxk, attnb);

  // 7) output projection + residual -> feats2
  gemm(wmma_gemm_kernel<false, true>, attnb, Wo_t, bo, feats, feats2, nullptr, BN_, D_, D_);

  // 8) LN(feats2) -> h_ln
  ln_kernel<<<BN_ / 8, 256, 0, stream>>>(feats2, ln_f_g, ln_f_b, h_ln, BN_);

  // 9) FFN
  gemm(wmma_gemm_kernel<true,  false>, h_ln, Wf1_t, b_ff1, nullptr, nullptr, ff1, BN_, FF_, D_);
  gemm(wmma_gemm_kernel<false, true>,  ff1,  Wf2_t, b_ff2, feats2, outp, nullptr, BN_, D_, FF_);
}
